// AdjacencyEstimator_32830730011154
// MI455X (gfx1250) — 8-bit matrix operand order repaired and checked by reading, compiled, not run on hardware
//
#include <hip/hip_runtime.h>
#include <stdint.h>

typedef int v8i __attribute__((ext_vector_type(8)));
typedef unsigned int uint32;

#define NLAB   32
#define DIM    96
#define SLICE  (DIM * DIM)
#define VOL    (DIM * DIM * DIM)
#define NBATCH 2
#define TOTAL  (NBATCH * VOL)
#define PL     104
#define ROWS   98
#define PLANE  (PL * ROWS)
#define CHUNKS (SLICE / 64)

__device__ __forceinline__ uint32 eqz01(uint32 x) {
  uint32 t = ((x & 0x7F7F7F7Fu) + 0x7F7F7F7Fu) | x;
  return ((t & 0x80808080u) ^ 0x80808080u) >> 7;
}

__device__ __forceinline__ uint32 funnel(uint32 hi, uint32 lo, int shbits) {
  return (uint32)((((unsigned long long)hi << 32) | lo) >> shbits);
}

__global__ __launch_bounds__(256) void convert_kernel(const int* __restrict__ t,
                                                      unsigned char* __restrict__ lab,
                                                      int* __restrict__ counts) {
  const int stride = gridDim.x * blockDim.x;
  const int gid = blockIdx.x * blockDim.x + threadIdx.x;
  for (int i = gid; i < TOTAL; i += stride) {
    __builtin_prefetch(t + i + 8192, 0, 1);
    lab[i] = (unsigned char)t[i];
  }
  if (gid < NLAB * NLAB) counts[gid] = 0;
}

__global__ __launch_bounds__(256) void adjacency_wmma_kernel(const unsigned char* __restrict__ lab,
                                                             int* __restrict__ counts) {
  __shared__ unsigned char tile[3 * PLANE];
  __shared__ int partial[NLAB * NLAB];

  const int tid = threadIdx.x;
  const int n = blockIdx.x / DIM;
  const int z = blockIdx.x % DIM;
  const unsigned char* base = lab + n * VOL;

  uint32* tile32 = (uint32*)tile;
  for (int i = tid; i < (3 * PLANE) / 4; i += 256) tile32[i] = 0xFFFFFFFFu;
  for (int i = tid; i < NLAB * NLAB; i += 256) partial[i] = 0;
  __syncthreads();
  for (int p = 0; p < 3; ++p) {
    const int zz = z + p - 1;
    if (zz < 0 || zz >= DIM) continue;
    const uint32* src = (const uint32*)(base + zz * SLICE);
    for (int i = tid; i < SLICE / 4; i += 256) {
      const int y = i / 24, xw = i - y * 24;
      *(uint32*)&tile[p * PLANE + (y + 1) * PL + 4 + xw * 4] = src[i];
    }
  }
  __syncthreads();

  const int lane = tid & 31;
  const int wave = tid >> 5;
  const int half = lane >> 4;
  const uint32 nsel = (uint32)(lane & 15);
  const uint32 splat0 = nsel * 0x01010101u;
  const uint32 splat1 = (nsel + 16u) * 0x01010101u;

  v8i acc00 = {}, acc01 = {}, acc10 = {}, acc11 = {};

  for (int c = wave; c < CHUNKS; c += 8) {
    const int v0 = c * 64;

    v8i A0, A1;
#pragma unroll
    for (int v = 0; v < 8; ++v) {
      const int K = ((v >> 2) << 5) + half * 16 + ((v & 3) << 2);
      const int vox = v0 + K;
      const int y = vox / DIM, x0 = vox - y * DIM;
      const uint32 w = *(const uint32*)&tile[PLANE + (y + 1) * PL + (x0 + 4)];
      A0[v] = (int)eqz01(w ^ splat0);
      A1[v] = (int)eqz01(w ^ splat1);
    }

    v8i B0, B1;
#pragma unroll
    for (int g = 0; g < 2; ++g) {
      const int vox0 = v0 + g * 32 + half * 16;
      const int y = vox0 / DIM, x0 = vox0 - y * DIM;
      const int px = x0 + 4;

      uint32 Z0[3][6], Z1[3][6];
#pragma unroll
      for (int r = 0; r < 3; ++r) {
#pragma unroll
        for (int w = 0; w < 6; ++w) {
          const int off = (y + r) * PL + px + (w - 1) * 4;
          uint32 a0 = 0, a1 = 0;
#pragma unroll
          for (int p = 0; p < 3; ++p) {
            const uint32 lw = *(const uint32*)&tile[p * PLANE + off];
            a0 += eqz01(lw ^ splat0);
            a1 += eqz01(lw ^ splat1);
          }
          Z0[r][w] = a0;
          Z1[r][w] = a1;
        }
      }
#pragma unroll
      for (int j = 0; j < 4; ++j) {
        uint32 s0 = 0, s1 = 0;
#pragma unroll
        for (int r = 0; r < 3; ++r) {
          s0 += Z0[r][j + 1] + funnel(Z0[r][j + 1], Z0[r][j], 24)
                             + funnel(Z0[r][j + 2], Z0[r][j + 1], 8);
          s1 += Z1[r][j + 1] + funnel(Z1[r][j + 1], Z1[r][j], 24)
                             + funnel(Z1[r][j + 2], Z1[r][j + 1], 8);
        }
        B0[g * 4 + j] = (int)s0;
        B1[g * 4 + j] = (int)s1;
      }
    }

    acc00 = __builtin_amdgcn_wmma_i32_16x16x64_iu8(false, A0, false, B0, acc00, false, false);
    acc01 = __builtin_amdgcn_wmma_i32_16x16x64_iu8(false, A0, false, B1, acc01, false, false);
    acc10 = __builtin_amdgcn_wmma_i32_16x16x64_iu8(false, A1, false, B0, acc10, false, false);
    acc11 = __builtin_amdgcn_wmma_i32_16x16x64_iu8(false, A1, false, B1, acc11, false, false);
  }

  const int Mbase = half * 8;
  const int Ncol = lane & 15;
#pragma unroll
  for (int r = 0; r < 8; ++r) {
    atomicAdd(&partial[(Mbase + r) * NLAB + Ncol],           acc00[r]);
    atomicAdd(&partial[(Mbase + r) * NLAB + Ncol + 16],      acc01[r]);
    atomicAdd(&partial[(Mbase + r + 16) * NLAB + Ncol],      acc10[r]);
    atomicAdd(&partial[(Mbase + r + 16) * NLAB + Ncol + 16], acc11[r]);
  }
  __syncthreads();
  for (int i = tid; i < NLAB * NLAB; i += 256) atomicAdd(&counts[i], partial[i]);
}

__global__ __launch_bounds__(256) void to_float_kernel(const int* __restrict__ counts,
                                                       float* __restrict__ out) {
  const int gid = blockIdx.x * blockDim.x + threadIdx.x;
  if (gid < NLAB * NLAB) out[gid] = (float)counts[gid];
}

extern "C" void kernel_launch(void* const* d_in, const int* in_sizes, int n_in,
                              void* d_out, int out_size, void* d_ws, size_t ws_size,
                              hipStream_t stream) {
  (void)in_sizes; (void)n_in; (void)out_size; (void)ws_size;
  const int* target = (const int*)d_in[0];
  unsigned char* lab8 = (unsigned char*)d_ws;
  int* counts = (int*)((char*)d_ws + TOTAL);

  convert_kernel<<<1024, 256, 0, stream>>>(target, lab8, counts);
  adjacency_wmma_kernel<<<NBATCH * DIM, 256, 0, stream>>>(lab8, counts);
  to_float_kernel<<<4, 256, 0, stream>>>(counts, (float*)d_out);
}
